// MultiHeadAttention_4363686773482
// MI455X (gfx1250) — matrix operand order repaired and checked by reading, not compiled, not run on hardware
//
#include <hip/hip_runtime.h>
#include <hip/hip_bf16.h>

// ---------------------------------------------------------------------------
// MHA for MI455X (gfx1250, wave32, WMMA).  f16 data / f32 accumulate.
//   D_MODEL=1024, H=16, D_K=64, B=2, S=2048  -> M = B*S = 4096 tokens
// ---------------------------------------------------------------------------

typedef _Float16 half8 __attribute__((ext_vector_type(8)));
typedef _Float16 v16h  __attribute__((ext_vector_type(16)));
typedef float    v8f   __attribute__((ext_vector_type(8)));

#define D_MODEL 1024
#define NHEADS  16
#define DK      64
#define BATCH   2
#define SEQ     2048
#define MTOK    (BATCH * SEQ)   // 4096

__device__ __forceinline__ v8f wmma_f16(v16h a, v16h b, v8f c) {
  // D = A(16x32 f16) * B(32x16 f16) + C(16x16 f32)
  return __builtin_amdgcn_wmma_f32_16x16x32_f16(
      false, a, false, b, (short)0, c, false, false);
}

// 16 contiguous halves -> B fragment (lane = column, K = koffB + [0..15])
__device__ __forceinline__ v16h ld_frag16(const _Float16* p) {
  half8 a = *(const half8*)p;
  half8 b = *(const half8*)(p + 8);
  return __builtin_shufflevector(a, b, 0,1,2,3,4,5,6,7,8,9,10,11,12,13,14,15);
}

// halves [p..p+7] and [p+16..p+23] -> A fragment (lane = row, koffA pre-added)
__device__ __forceinline__ v16h ld_frag8x2(const _Float16* p) {
  half8 a = *(const half8*)p;
  half8 b = *(const half8*)(p + 16);
  return __builtin_shufflevector(a, b, 0,1,2,3,4,5,6,7,8,9,10,11,12,13,14,15);
}

// ------------------------------ conversions --------------------------------
__global__ void cvt_f32_to_f16(const float* __restrict__ in,
                               _Float16* __restrict__ out, int n) {
  int i = blockIdx.x * blockDim.x + threadIdx.x;
  if (i < n) out[i] = (_Float16)in[i];
}

// W stored [in,out] row-major fp32 -> WT [out,in] row-major f16
__global__ void cvt_transpose_f16(const float* __restrict__ W,
                                  _Float16* __restrict__ WT, int n) {
  int i = blockIdx.x * blockDim.x + threadIdx.x;
  if (i >= n) return;
  int o  = i >> 10;          // / D_MODEL
  int in = i & (D_MODEL - 1);
  WT[i] = (_Float16)W[in * D_MODEL + o];
}

// ------------------------------ tiled GEMM ---------------------------------
// C[M,N] = A[M,K](f16 rm) * Bt[N,K](f16 rm, i.e. B^T) + bias
// one wave per 16x64 tile; K stepped by 32.
// mode 0: fp32 out[row*N+col]
// mode 1: f16  out[((b*16+h)*SEQ+s)*64+dk]          (Q/K head-split)
// mode 2: f16  out[((b*16+h)*64+dk)*SEQ+s]          (V head-split, transposed)
__global__ __launch_bounds__(128)
void gemm_wmma(const _Float16* __restrict__ A, const _Float16* __restrict__ Bt,
               const float* __restrict__ bias, void* __restrict__ outp,
               int M, int N, int K, int mode) {
  const int warp = threadIdx.x >> 5;
  const int lane = threadIdx.x & 31;
  const int tile = blockIdx.x * (blockDim.x >> 5) + warp;
  const int tilesN = N >> 6;
  const int tm = tile / tilesN;
  const int tn = tile - tm * tilesN;
  if (tm * 16 >= M) return;
  const int m0 = tm << 4, n0 = tn << 6;
  const int l16 = lane & 15;
  const int hi  = (lane >> 4) & 1;
  const int koffA = hi ? 16 : 0;
  const int koffB = hi ? 16 : 0;

  const _Float16* arow = A + (size_t)(m0 + l16) * K + koffA;
  const _Float16* b0   = Bt + (size_t)(n0 + l16) * K + koffB;
  const size_t bstep = (size_t)16 * K;

  v8f acc[4] = {};
  for (int kb = 0; kb < K; kb += 32) {
    v16h af = ld_frag16(arow + kb);
    acc[0] = wmma_f16(af, ld_frag16(b0 + kb),             acc[0]);
    acc[1] = wmma_f16(af, ld_frag16(b0 + kb + bstep),     acc[1]);
    acc[2] = wmma_f16(af, ld_frag16(b0 + kb + 2 * bstep), acc[2]);
    acc[3] = wmma_f16(af, ld_frag16(b0 + kb + 3 * bstep), acc[3]);
  }

#pragma unroll
  for (int nt = 0; nt < 4; ++nt) {
    const int col = n0 + nt * 16 + l16;
    const float bv = bias[col];
#pragma unroll
    for (int r = 0; r < 8; ++r) {
      const int row = m0 + r + (hi << 3);
      const float val = acc[nt][r] + bv;
      if (mode == 0) {
        ((float*)outp)[(size_t)row * N + col] = val;
      } else {
        const int b = row >> 11, s = row & (SEQ - 1);
        const int h = col >> 6,  dk = col & (DK - 1);
        _Float16* o = (_Float16*)outp;
        if (mode == 1)
          o[((size_t)((b << 4) + h) * SEQ + s) * DK + dk] = (_Float16)val;
        else
          o[((size_t)((b << 4) + h) * DK + dk) * SEQ + s] = (_Float16)val;
      }
    }
  }
}

// --------------------------- flash attention -------------------------------
// one wave per (b,h, 16-query tile).  Q/K: [bh, s, dk] f16.  Vt: [bh, dk, s].
// out: [b, s, h*64+dk] f16  (heads concatenated -> A matrix of final proj)
__global__ __launch_bounds__(128)
void attn_wmma(const _Float16* __restrict__ Qh, const _Float16* __restrict__ Kh,
               const _Float16* __restrict__ Vt, _Float16* __restrict__ out) {
  __shared__ _Float16 lds[4][16 * 32];   // per-wave 16x32 P-tile staging
  const int warp = threadIdx.x >> 5;
  const int lane = threadIdx.x & 31;
  const int w = blockIdx.x * 4 + warp;
  const int qtile = w & (SEQ / 16 - 1);   // 0..127
  const int bh    = w >> 7;               // 0..31
  if (bh >= BATCH * NHEADS) return;
  const int q0  = qtile << 4;
  const int l16 = lane & 15;
  const int hi  = (lane >> 4) & 1;
  const int koffA = hi ? 16 : 0;
  const int koffB = hi ? 16 : 0;

  const _Float16* Qb = Qh + (size_t)bh * SEQ * DK;
  const _Float16* Kb = Kh + (size_t)bh * SEQ * DK;
  const _Float16* Vb = Vt + (size_t)bh * DK * SEQ;

  // Q A-fragments for full dk=64 (two k-steps), loaded once.
  const _Float16* qp = Qb + (size_t)(q0 + l16) * DK + koffA;
  const v16h qf0 = ld_frag16(qp);
  const v16h qf1 = ld_frag16(qp + 32);

  float mrow[8], lrow[8];
  v8f acc[4] = {};
#pragma unroll
  for (int r = 0; r < 8; ++r) { mrow[r] = -__builtin_inff(); lrow[r] = 0.f; }

  const float scale = 0.125f;   // 1/sqrt(64)

  for (int kb = 0; kb <= q0 + 15; kb += 32) {   // causal: skip future blocks
    // ---- scores: two 16x16 tiles (keys kb..+15 and kb+16..+31) ----
    v8f s0 = {}, s1 = {};
    {
      const _Float16* kp0 = Kb + (size_t)(kb + l16) * DK + koffB;
      s0 = wmma_f16(qf0, ld_frag16(kp0),      s0);
      s0 = wmma_f16(qf1, ld_frag16(kp0 + 32), s0);
      const _Float16* kp1 = Kb + (size_t)(kb + 16 + l16) * DK + koffB;
      s1 = wmma_f16(qf0, ld_frag16(kp1),      s1);
      s1 = wmma_f16(qf1, ld_frag16(kp1 + 32), s1);
    }
    const int k0 = kb + l16, k1 = kb + 16 + l16;

    // ---- online softmax update, stage P tile to LDS ----
#pragma unroll
    for (int r = 0; r < 8; ++r) {
      const int q = q0 + r + (hi << 3);
      float sc0 = s0[r] * scale;
      float sc1 = s1[r] * scale;
      if (k0 > q) sc0 = -__builtin_inff();
      if (k1 > q) sc1 = -__builtin_inff();
      float vmax = fmaxf(sc0, sc1);
#pragma unroll
      for (int off = 8; off; off >>= 1)
        vmax = fmaxf(vmax, __shfl_xor(vmax, off, 32));
      const float mn = fmaxf(mrow[r], vmax);
      const float alpha = __expf(mrow[r] - mn);
      const float p0 = __expf(sc0 - mn);
      const float p1 = __expf(sc1 - mn);
      float rs = p0 + p1;
#pragma unroll
      for (int off = 8; off; off >>= 1)
        rs += __shfl_xor(rs, off, 32);
      lrow[r] = lrow[r] * alpha + rs;
      mrow[r] = mn;
#pragma unroll
      for (int nt = 0; nt < 4; ++nt) acc[nt][r] *= alpha;
      const int prow = r + (hi << 3);
      lds[warp][prow * 32 + l16]      = (_Float16)p0;
      lds[warp][prow * 32 + 16 + l16] = (_Float16)p1;
    }
    asm volatile("s_wait_dscnt 0" ::: "memory");

    // ---- re-layout P (D-frag in LDS) -> A-frag, accumulate P x V ----
    const v16h pf = ld_frag16(&lds[warp][l16 * 32 + koffA]);
#pragma unroll
    for (int nt = 0; nt < 4; ++nt) {
      const _Float16* vp = Vb + (size_t)(nt * 16 + l16) * SEQ + kb + koffB;
      acc[nt] = wmma_f16(pf, ld_frag16(vp), acc[nt]);
    }
  }

  // ---- normalize and store (concat-heads layout) ----
  const int b = bh >> 4, h = bh & 15;
#pragma unroll
  for (int r = 0; r < 8; ++r) {
    const float inv = 1.0f / lrow[r];
    const int q = q0 + r + (hi << 3);
    _Float16* orow = out + ((size_t)(b * SEQ + q) * D_MODEL) + h * DK;
#pragma unroll
    for (int nt = 0; nt < 4; ++nt)
      orow[nt * 16 + l16] = (_Float16)(acc[nt][r] * inv);
  }
}

// ------------------------------- launcher ----------------------------------
extern "C" void kernel_launch(void* const* d_in, const int* in_sizes, int n_in,
                              void* d_out, int out_size, void* d_ws, size_t ws_size,
                              hipStream_t stream) {
  const float* x  = (const float*)d_in[0];
  // d_in[1] = mask (causal triu) -- implemented analytically in the kernel
  const float* Wq = (const float*)d_in[2];
  const float* bq = (const float*)d_in[3];
  const float* Wk = (const float*)d_in[4];
  const float* bk = (const float*)d_in[5];
  const float* Wv = (const float*)d_in[6];
  const float* bv = (const float*)d_in[7];
  const float* Wo = (const float*)d_in[8];
  const float* bo = (const float*)d_in[9];
  float* out = (float*)d_out;

  const size_t MB = 1u << 20;
  char* ws = (char*)d_ws;
  _Float16* xh   = (_Float16*)(ws);             //  8 MB  x as f16 [4096,1024]
  _Float16* WqT  = (_Float16*)(ws +  8 * MB);   //  2 MB
  _Float16* WkT  = (_Float16*)(ws + 10 * MB);
  _Float16* WvT  = (_Float16*)(ws + 12 * MB);
  _Float16* WoT  = (_Float16*)(ws + 14 * MB);
  _Float16* Qh   = (_Float16*)(ws + 16 * MB);   //  8 MB [b,h,s,dk]
  _Float16* Kh   = (_Float16*)(ws + 24 * MB);   //  8 MB [b,h,s,dk]
  _Float16* Vt   = (_Float16*)(ws + 32 * MB);   //  8 MB [b,h,dk,s]
  _Float16* Ao   = (_Float16*)(ws + 40 * MB);   //  8 MB attn out [4096,1024]

  const int nX = MTOK * D_MODEL;      // 4194304
  const int nW = D_MODEL * D_MODEL;   // 1048576
  cvt_f32_to_f16<<<(nX + 255) / 256, 256, 0, stream>>>(x, xh, nX);
  cvt_transpose_f16<<<(nW + 255) / 256, 256, 0, stream>>>(Wq, WqT, nW);
  cvt_transpose_f16<<<(nW + 255) / 256, 256, 0, stream>>>(Wk, WkT, nW);
  cvt_transpose_f16<<<(nW + 255) / 256, 256, 0, stream>>>(Wv, WvT, nW);
  cvt_transpose_f16<<<(nW + 255) / 256, 256, 0, stream>>>(Wo, WoT, nW);

  // (M/16)*(N/64) = 256*16 = 4096 wave-tiles, 4 waves/block -> 1024 blocks
  const int gemmBlocks = (MTOK / 16) * (D_MODEL / 64) / 4;
  gemm_wmma<<<gemmBlocks, 128, 0, stream>>>(xh, WqT, bq, Qh, MTOK, D_MODEL, D_MODEL, 1);
  gemm_wmma<<<gemmBlocks, 128, 0, stream>>>(xh, WkT, bk, Kh, MTOK, D_MODEL, D_MODEL, 1);
  gemm_wmma<<<gemmBlocks, 128, 0, stream>>>(xh, WvT, bv, Vt, MTOK, D_MODEL, D_MODEL, 2);

  // B*H*(S/16) = 4096 waves, 4 waves/block -> 1024 blocks
  attn_wmma<<<(BATCH * NHEADS * (SEQ / 16)) / 4, 128, 0, stream>>>(Qh, Kh, Vt, Ao);

  gemm_wmma<<<gemmBlocks, 128, 0, stream>>>(Ao, WoT, bo, out, MTOK, D_MODEL, D_MODEL, 0);
}
